// MultiHeadAttention_7713761263676
// MI455X (gfx1250) — matrix operand order repaired and checked by reading, not compiled, not run on hardware
//
#include <hip/hip_runtime.h>

// ---------------------------------------------------------------------------
// MHA forward for MI455X (gfx1250, wave32, WMMA f32_16x16x32_f16).
// fp32 accumulation, f16 operands (flash-attention numerics).
// Projections: 16(M) x 64(N) tile per wave; A converted once per K-step and
// reused by 4 WMMAs; all B tiles held in distinct registers so their loads
// issue as one clause and overlap with WMMA execution.
// ---------------------------------------------------------------------------

typedef __attribute__((ext_vector_type(16))) _Float16 v16h;
typedef __attribute__((ext_vector_type(8)))  float    v8f;

constexpr int Bc  = 2;
constexpr int Tc  = 4096;
constexpr int Dc  = 512;
constexpr int Hc  = 8;
constexpr int DKc = 64;   // D / H

#define WMMA_F16(a, b, c) \
  __builtin_amdgcn_wmma_f32_16x16x32_f16(false, (a), false, (b), (short)0, (c), false, false)

// ---------------------------------------------------------------- fp32 -> f16
__global__ void cvt_f32_f16_kernel(const float* __restrict__ src,
                                   _Float16* __restrict__ dst, int n) {
  int i = blockIdx.x * blockDim.x + threadIdx.x;
  if (i < n) dst[i] = (_Float16)src[i];
}

// ------------------------------------------------- fused QKV projection GEMM
// One wave computes a 16x64 tile of  y = x @ W^T + b  (M over B*T, N = one head).
// blockIdx.z selects q / k / v.  q,k -> [B,H,T,DK] f16 ; v -> [B,H,DK,T] f16.
__global__ __launch_bounds__(32) void qkv_proj_kernel(
    const float* __restrict__ xq, const float* __restrict__ xk,
    const float* __restrict__ xv,
    const _Float16* __restrict__ Wq, const _Float16* __restrict__ Wk,
    const _Float16* __restrict__ Wv,
    const float* __restrict__ bq, const float* __restrict__ bk,
    const float* __restrict__ bv,
    _Float16* __restrict__ Qh, _Float16* __restrict__ Kh,
    _Float16* __restrict__ Vt) {
  const int lane = threadIdx.x;
  const int half = lane >> 4;       // 0 / 1
  const int l16  = lane & 15;
  const int kbh  = half * 8;        // A-operand K sub-offset
  const int row0 = blockIdx.x * 16; // over B*T
  const int h    = blockIdx.y;      // head index; N range = [h*64, h*64+64)
  const int col0 = h * DKc;
  const int sel  = blockIdx.z;      // 0=q 1=k 2=v

  const float*    X = (sel == 0) ? xq : (sel == 1) ? xk : xv;
  const _Float16* W = (sel == 0) ? Wq : (sel == 1) ? Wk : Wv;
  const float*    bias = (sel == 0) ? bq : (sel == 1) ? bk : bv;

  v8f c0 = {}, c1 = {}, c2 = {}, c3 = {};
  const int arow = row0 + l16;
  for (int kk = 0; kk < Dc; kk += 32) {
    // B tiles first: 8 independent b128 loads issue as one clause.
    v16h b0, b1, b2, b3;
    const _Float16* wp = W + (col0 + l16) * Dc + kk + half * 16;
#pragma unroll
    for (int j = 0; j < 16; ++j) {
      b0[j] = wp[j];
      b1[j] = wp[16 * Dc + j];
      b2[j] = wp[32 * Dc + j];
      b3[j] = wp[48 * Dc + j];
    }
    v16h a;
    const float* xp = X + arow * Dc + kk + half * 16;
#pragma unroll
    for (int i = 0; i < 8; ++i) {
      a[i]     = (_Float16)xp[i];
      a[i + 8] = (_Float16)xp[8 + i];
    }
    c0 = WMMA_F16(a, b0, c0);
    c1 = WMMA_F16(a, b1, c1);
    c2 = WMMA_F16(a, b2, c2);
    c3 = WMMA_F16(a, b3, c3);
  }

  const float b0s = bias[col0 + l16];
  const float b1s = bias[col0 + 16 + l16];
  const float b2s = bias[col0 + 32 + l16];
  const float b3s = bias[col0 + 48 + l16];
#pragma unroll
  for (int r = 0; r < 8; ++r) {
    const int grow = row0 + r + 8 * half;    // global row in B*T
    const int bidx = grow >> 12;             // / Tc
    const int t    = grow & (Tc - 1);
    const _Float16 v0 = (_Float16)(c0[r] + b0s);
    const _Float16 v1 = (_Float16)(c1[r] + b1s);
    const _Float16 v2 = (_Float16)(c2[r] + b2s);
    const _Float16 v3 = (_Float16)(c3[r] + b3s);
    if (sel == 2) {
      // transposed: [B,H,DK,T]
      _Float16* vp = Vt + (((long)(bidx * Hc + h) * DKc + l16) * Tc) + t;
      vp[0 * 16 * Tc] = v0;
      vp[1 * 16 * Tc] = v1;
      vp[2 * 16 * Tc] = v2;
      vp[3 * 16 * Tc] = v3;
    } else {
      _Float16* op = ((sel == 0) ? Qh : Kh) +
                     (((long)(bidx * Hc + h) * Tc + t) * DKc) + l16;
      op[0]  = v0;
      op[16] = v1;
      op[32] = v2;
      op[48] = v3;
    }
  }
}

// ------------------------------------------------------- causal flash attention
// One wave per (b*h, 16-row q tile).  K/V streamed in 32-column blocks.
__global__ __launch_bounds__(32) void flash_attn_kernel(
    const _Float16* __restrict__ Qh,   // [B*H, T, DK]
    const _Float16* __restrict__ Kh,   // [B*H, T, DK]
    const _Float16* __restrict__ Vt,   // [B*H, DK, T]
    _Float16* __restrict__ Oh) {       // [B, T, D]  (col = h*DK + d)
  const int lane = threadIdx.x;
  const int half = lane >> 4;
  const int l16  = lane & 15;
  const int kbh  = half * 8;
  const int q0   = blockIdx.x * 16;
  const int bh   = blockIdx.y;
  const int b    = bh >> 3;            // / Hc
  const int h    = bh & (Hc - 1);

  const _Float16* Qb = Qh + (long)bh * Tc * DKc;
  const _Float16* Kb = Kh + (long)bh * Tc * DKc;
  const _Float16* Vb = Vt + (long)bh * DKc * Tc;

  // Q tile A-operands for d = 0..31 (qa0) and d = 32..63 (qa1): loaded once.
  v16h qa0, qa1;
  {
    const _Float16* qp = Qb + (q0 + l16) * DKc;
#pragma unroll
    for (int i = 0; i < 8; ++i) {
      qa0[i]     = qp[half * 16 + i];
      qa0[i + 8] = qp[half * 16 + 8 + i];
      qa1[i]     = qp[32 + half * 16 + i];
      qa1[i + 8] = qp[32 + half * 16 + 8 + i];
    }
  }

  v8f O0 = {}, O1 = {}, O2 = {}, O3 = {};
  float mrow[8], lsum[8];
#pragma unroll
  for (int r = 0; r < 8; ++r) { mrow[r] = -1.0e30f; lsum[r] = 0.0f; }

  __shared__ _Float16 Pl[16][32];
  const float scale = 0.125f;  // 1/sqrt(DK)

  for (int kb = 0; kb < q0 + 16; kb += 32) {
    // ---- S = Q K^T for columns [kb, kb+32) : two 16x16 tiles, K(d)=64 ----
    // All four K-operand tiles loaded into distinct regs -> one load clause.
    v8f s0 = {}, s1 = {};
    {
      v16h ka0, ka1, kc0, kc1;
      const _Float16* kp0 = Kb + (kb + l16) * DKc + half * 16;
      const _Float16* kp1 = Kb + (kb + 16 + l16) * DKc + half * 16;
#pragma unroll
      for (int j = 0; j < 16; ++j) {
        ka0[j] = kp0[j];
        ka1[j] = kp1[j];
        kc0[j] = kp0[32 + j];
        kc1[j] = kp1[32 + j];
      }
      s0 = WMMA_F16(qa0, ka0, s0);
      s1 = WMMA_F16(qa0, ka1, s1);
      s0 = WMMA_F16(qa1, kc0, s0);
      s1 = WMMA_F16(qa1, kc1, s1);
    }

    // ---- online softmax (rows live in 16-lane half-groups) ----
    const bool diag = (kb + 32 > q0);
    float p0[8], p1[8], corr[8];
#pragma unroll
    for (int r = 0; r < 8; ++r) {
      const int qrow = q0 + r + 8 * half;
      float v0 = s0[r] * scale;
      float v1 = s1[r] * scale;
      if (diag) {
        if (kb + l16 > qrow)      v0 = -3.0e38f;
        if (kb + 16 + l16 > qrow) v1 = -3.0e38f;
      }
      float bm = fmaxf(v0, v1);
      bm = fmaxf(bm, __shfl_xor(bm, 1, 32));
      bm = fmaxf(bm, __shfl_xor(bm, 2, 32));
      bm = fmaxf(bm, __shfl_xor(bm, 4, 32));
      bm = fmaxf(bm, __shfl_xor(bm, 8, 32));
      const float mn = fmaxf(mrow[r], bm);
      const float e0 = __expf(v0 - mn);
      const float e1 = __expf(v1 - mn);
      float rs = e0 + e1;
      rs += __shfl_xor(rs, 1, 32);
      rs += __shfl_xor(rs, 2, 32);
      rs += __shfl_xor(rs, 4, 32);
      rs += __shfl_xor(rs, 8, 32);
      const float cf = __expf(mrow[r] - mn);
      lsum[r] = lsum[r] * cf + rs;
      mrow[r] = mn;
      corr[r] = cf;
      p0[r] = e0;
      p1[r] = e1;
    }

    // ---- V tiles: issue 8 b128 loads early so they overlap softmax VALU ----
    v16h vb0, vb1, vb2, vb3;
    {
      const _Float16* vp = Vb + l16 * Tc + kb + half * 16;
#pragma unroll
      for (int j = 0; j < 16; ++j) {
        vb0[j] = vp[j];
        vb1[j] = vp[16 * Tc + j];
        vb2[j] = vp[32 * Tc + j];
        vb3[j] = vp[48 * Tc + j];
      }
    }

#pragma unroll
    for (int r = 0; r < 8; ++r) {
      O0[r] *= corr[r]; O1[r] *= corr[r]; O2[r] *= corr[r]; O3[r] *= corr[r];
    }

    // ---- C-layout P -> A-layout via LDS (wave-private, DS ops in order) ----
#pragma unroll
    for (int r = 0; r < 8; ++r) {
      Pl[r + 8 * half][l16]      = (_Float16)p0[r];
      Pl[r + 8 * half][16 + l16] = (_Float16)p1[r];
    }
    asm volatile("s_wait_dscnt 0" ::: "memory");
    v16h pa;
#pragma unroll
    for (int i = 0; i < 8; ++i) {
      pa[i]     = Pl[l16][half * 16 + i];
      pa[i + 8] = Pl[l16][half * 16 + 8 + i];
    }

    // ---- O += P @ V  (4 d-tiles of 16, V pre-transposed -> contiguous B) ----
    O0 = WMMA_F16(pa, vb0, O0);
    O1 = WMMA_F16(pa, vb1, O1);
    O2 = WMMA_F16(pa, vb2, O2);
    O3 = WMMA_F16(pa, vb3, O3);
  }

  // ---- normalize and store f16 [B,T,D] ----
#pragma unroll
  for (int r = 0; r < 8; ++r) {
    const float inv  = 1.0f / lsum[r];
    const int   qrow = q0 + r + 8 * half;
    _Float16* op = Oh + ((long)b * Tc + qrow) * Dc + h * DKc + l16;
    op[0]  = (_Float16)(O0[r] * inv);
    op[16] = (_Float16)(O1[r] * inv);
    op[32] = (_Float16)(O2[r] * inv);
    op[48] = (_Float16)(O3[r] * inv);
  }
}

// ----------------------------------------------------- output projection GEMM
// out = A16 @ Wo^T + bo (fp32 store), 16x64 tile per wave (A reused 4x).
__global__ __launch_bounds__(32) void out_proj_kernel(
    const _Float16* __restrict__ A, const _Float16* __restrict__ W,
    const float* __restrict__ bias, float* __restrict__ out) {
  const int lane = threadIdx.x;
  const int half = lane >> 4;
  const int l16  = lane & 15;
  const int kbh  = half * 8;
  const int row0 = blockIdx.x * 16;
  const int col0 = blockIdx.y * 64;

  v8f c0 = {}, c1 = {}, c2 = {}, c3 = {};
  for (int kk = 0; kk < Dc; kk += 32) {
    v16h b0, b1, b2, b3;
    const _Float16* wp = W + (col0 + l16) * Dc + kk + half * 16;
#pragma unroll
    for (int j = 0; j < 16; ++j) {
      b0[j] = wp[j];
      b1[j] = wp[16 * Dc + j];
      b2[j] = wp[32 * Dc + j];
      b3[j] = wp[48 * Dc + j];
    }
    v16h a;
    const _Float16* ap = A + (row0 + l16) * Dc + kk;
#pragma unroll
    for (int i = 0; i < 8; ++i) {
      a[i]     = ap[half * 16 + i];
      a[i + 8] = ap[half * 16 + 8 + i];
    }
    c0 = WMMA_F16(a, b0, c0);
    c1 = WMMA_F16(a, b1, c1);
    c2 = WMMA_F16(a, b2, c2);
    c3 = WMMA_F16(a, b3, c3);
  }
  const float b0s = bias[col0 + l16];
  const float b1s = bias[col0 + 16 + l16];
  const float b2s = bias[col0 + 32 + l16];
  const float b3s = bias[col0 + 48 + l16];
#pragma unroll
  for (int r = 0; r < 8; ++r) {
    float* op = out + (long)(row0 + r + 8 * half) * Dc + col0 + l16;
    op[0]  = c0[r] + b0s;
    op[16] = c1[r] + b1s;
    op[32] = c2[r] + b2s;
    op[48] = c3[r] + b3s;
  }
}

// ---------------------------------------------------------------------------
extern "C" void kernel_launch(void* const* d_in, const int* in_sizes, int n_in,
                              void* d_out, int out_size, void* d_ws, size_t ws_size,
                              hipStream_t stream) {
  const float* q  = (const float*)d_in[0];
  const float* k  = (const float*)d_in[1];
  const float* v  = (const float*)d_in[2];
  // d_in[3] = mask : causality implemented analytically
  const float* Wq = (const float*)d_in[4];
  const float* bq = (const float*)d_in[5];
  const float* Wk = (const float*)d_in[6];
  const float* bk = (const float*)d_in[7];
  const float* Wv = (const float*)d_in[8];
  const float* bv = (const float*)d_in[9];
  const float* Wo = (const float*)d_in[10];
  const float* bo = (const float*)d_in[11];
  float* out = (float*)d_out;

  const int WN  = Dc * Dc;              // 262144 elements per weight
  const int PHN = Bc * Hc * Tc * DKc;   // 4194304 per projected tensor

  _Float16* ws   = (_Float16*)d_ws;
  _Float16* Wq16 = ws;
  _Float16* Wk16 = Wq16 + WN;
  _Float16* Wv16 = Wk16 + WN;
  _Float16* Wo16 = Wv16 + WN;
  _Float16* Qh   = Wo16 + WN;
  _Float16* Kh   = Qh + PHN;
  _Float16* Vt   = Kh + PHN;
  _Float16* Oh   = Vt + PHN;            // [B,T,D] f16

  // 1) weights -> f16
  dim3 cb(256);
  dim3 cg((WN + 255) / 256);
  cvt_f32_f16_kernel<<<cg, cb, 0, stream>>>(Wq, Wq16, WN);
  cvt_f32_f16_kernel<<<cg, cb, 0, stream>>>(Wk, Wk16, WN);
  cvt_f32_f16_kernel<<<cg, cb, 0, stream>>>(Wv, Wv16, WN);
  cvt_f32_f16_kernel<<<cg, cb, 0, stream>>>(Wo, Wo16, WN);

  // 2) QKV projections (one wave = one 16x64 tile = one head slice)
  dim3 pg((Bc * Tc) / 16, Hc, 3);
  qkv_proj_kernel<<<pg, dim3(32), 0, stream>>>(q, k, v, Wq16, Wk16, Wv16,
                                               bq, bk, bv, Qh, Kh, Vt);

  // 3) causal flash attention
  dim3 fg(Tc / 16, Bc * Hc);
  flash_attn_kernel<<<fg, dim3(32), 0, stream>>>(Qh, Kh, Vt, Oh);

  // 4) output projection (fp32 result, 16x64 tiles)
  dim3 og((Bc * Tc) / 16, Dc / 64);
  out_proj_kernel<<<og, dim3(32), 0, stream>>>(Oh, Wo16, bo, out);
}
